// RecurrentResidualColumnViT_81406810128590
// MI455X (gfx1250) — compile-verified, matrix operand order checked by reading, not run on hardware
//
#include <hip/hip_runtime.h>
#include <hip/hip_bf16.h>

// ---------------- types / helpers ----------------
typedef __attribute__((ext_vector_type(16))) __bf16 bf16x16;
typedef __attribute__((ext_vector_type(8)))  float  f32x8;

union FragU { bf16x16 v; uint4 q[2]; unsigned short s[16]; };

__device__ __forceinline__ unsigned short f2bf(float f) {
  unsigned int u = __float_as_uint(f);
  u += 0x7fffu + ((u >> 16) & 1u);
  return (unsigned short)(u >> 16);
}

__device__ __forceinline__ float cmodf(float x) {
  float y = x + 63.5f;
  y -= 127.0f * floorf(y * (1.0f / 127.0f));
  return y - 63.5f;
}

#define WQ        0.04947389218252431f   // 2*pi/127
#define INV_HQ    0.015748031496062992f  // 1/63.5
#define LN_EPS    1e-5f

#define LDX 132   // xf32 row stride (f32)
#define LDA 136   // abf row stride (bf16)
#define LDQ 392   // qkv row stride (bf16)
#define LDF 72    // conv feature chunk stride (bf16)
#define LDH 136   // ffn hidden chunk stride (bf16)

struct KParams {
  const float *A, *b, *conv_b, *pos;
  const float *ln1_s[2], *ln1_b[2], *qkv_bias[2], *out_bias[2];
  const float *ln2_s[2], *ln2_b[2], *ff1_bias[2], *ff2_bias[2];
  const float *lnf_s, *lnf_b, *head_w, *head_b;
  const unsigned short *conv_wb;
  const unsigned short *qkv_wb[2], *out_wb[2], *ff1_wb[2], *ff2_wb[2];
  float *out;
};

// one 16-row strip (rows row0..row0+15) x 128 cols (8 tiles), accumulate over nK32 k-tiles of 32
__device__ __forceinline__ void gemm_strip(f32x8 acc[8],
                                           const unsigned short* aBuf, int lda, int aRow0, int aK0,
                                           const unsigned short* bW, int ldb, int bN0, int bK0,
                                           int nK32, int lane) {
  const int m = lane & 15;
  const int half8 = (lane >> 4) << 3;
  for (int kk = 0; kk < nK32; ++kk) {
    FragU fa;
    const uint4* pa = (const uint4*)(aBuf + (aRow0 + m) * lda + aK0 + kk * 32 + half8);
    fa.q[0] = pa[0]; fa.q[1] = pa[2];
#pragma unroll
    for (int tc = 0; tc < 8; ++tc) {
      FragU fb;
      const uint4* pb = (const uint4*)(bW + (size_t)(bN0 + tc * 16 + m) * ldb + bK0 + kk * 32 + half8);
      fb.q[0] = pb[0]; fb.q[1] = pb[2];
      acc[tc] = __builtin_amdgcn_wmma_f32_16x16x32_bf16(false, fa.v, false, fb.v,
                                                        (short)0, acc[tc], false, false);
    }
  }
}

__device__ __forceinline__ float featval(int c, float a, float bm, float dm, float rm, float pn,
                                         float sma, float sci, float onc, float offc, float cur) {
  if (c == 0)  return cmodf(a) * INV_HQ;
  if (c == 1)  return __sinf(WQ * a);
  if (c == 2)  return __cosf(WQ * a);
  if (c == 3)  return cmodf(bm) * INV_HQ;
  if (c == 4)  return __sinf(WQ * bm);
  if (c == 5)  return __cosf(WQ * bm);
  if (c == 6)  return rm * INV_HQ;
  if (c == 7)  return __sinf(WQ * rm);
  if (c == 8)  return __cosf(WQ * rm);
  if (c == 9)  return pn;
  if (c == 10) return cmodf(a) * INV_HQ * rm * INV_HQ;
  if (c == 11) return cmodf(a) * INV_HQ * pn;
  if (c == 12) return sma;
  if (c == 13) return sci;
  if (c == 14) return cur;
  if (c == 15) return cur - onc;
  if (c == 16) return cur - offc;
  float on_r  = cmodf(dm - a * (1.0f - pn));
  float off_r = cmodf(dm + a * pn);
  if (c == 17) return on_r * INV_HQ;
  if (c == 18) return off_r * INV_HQ;
  return (on_r - off_r) * INV_HQ;
}

__global__ void f2bf_kernel(const float* __restrict__ s, unsigned short* __restrict__ d, int n) {
  int i = blockIdx.x * blockDim.x + threadIdx.x;
  if (i < n) d[i] = f2bf(s[i]);
}

// ---------------- main persistent kernel: 8 batch items / WG, 6 recurrent steps ----------------
__global__ __launch_bounds__(256, 1) void vit_kernel(KParams P) {
  extern __shared__ unsigned char smem[];
  float* A_l   = (float*)smem;            // [8][64][16]
  float* b_l   = A_l  + 8192;             // [8][64]
  float* z_l   = b_l  + 512;              // [128]
  float* p_l   = z_l  + 128;              // [128]
  float* d_l   = p_l  + 128;              // [8][64]
  float* r_l   = d_l  + 512;              // [8][64]
  float* sma_l = r_l  + 512;              // [128]
  float* sci_l = sma_l + 128;
  float* onc_l = sci_l + 128;
  float* offc_l = onc_l + 128;
  float* cur_l = offc_l + 128;            // [8] (+pad)
  float* xf32  = cur_l + 16;              // [128][LDX]
  unsigned short* abf  = (unsigned short*)(xf32 + 128 * LDX); // [128][LDA]
  unsigned short* qkvb = abf  + 128 * LDA;                    // [128][LDQ]
  unsigned short* scr  = qkvb + 128 * LDQ;                    // shared: fbuf / hbf / scbuf

  const int tid  = threadIdx.x;
  const int lane = tid & 31;
  const int wave = tid >> 5;
  const int wg   = blockIdx.x;
  const int nn   = lane & 15;
  const int mbase = (lane >> 4) << 3;
  const int half8 = mbase;

  // load A, b tiles; init z
  for (int i = tid; i < 8192; i += 256) A_l[i] = P.A[(size_t)wg * 8192 + i];
  for (int i = tid; i < 512;  i += 256) b_l[i] = P.b[(size_t)wg * 512 + i];
  if (tid < 128) z_l[tid] = 0.0f;
  __syncthreads();

  float* res_out = P.out + 65536;

  for (int t = 0; t < 6; ++t) {
    // ---- p = sigmoid(z) ----
    if (tid < 128) p_l[tid] = 1.0f / (1.0f + __expf(-z_l[tid]));
    __syncthreads();

    // ---- d = b - A p ; r = cmod(d) ; emit residuals ----
    for (int idx = tid; idx < 512; idx += 256) {
      int bi = idx >> 6, m = idx & 63;
      float acc = 0.0f;
#pragma unroll
      for (int n = 0; n < 16; ++n) acc += A_l[idx * 16 + n] * p_l[bi * 16 + n];
      float dd = b_l[idx] - acc;
      float rr = cmodf(dd);
      d_l[idx] = dd; r_l[idx] = rr;
      res_out[((size_t)(wg * 8 + bi) * 6 + t) * 64 + m] = rr;
    }
    __syncthreads();

    // ---- per-(b,n) stats ----
    if (tid < 128) {
      int bi = tid >> 4;
      float pn = p_l[tid];
      float sa = 0, sc2 = 0, son = 0, soff = 0;
      for (int m = 0; m < 64; ++m) {
        float a  = A_l[(bi * 64 + m) * 16 + (tid & 15)];
        float bm = b_l[bi * 64 + m];
        float dm = d_l[bi * 64 + m];
        float sr = cmodf(bm - a);
        sa  += fabsf(sr);
        sc2 += 1.0f - __cosf(WQ * sr);
        son  += 1.0f - __cosf(WQ * cmodf(dm - a * (1.0f - pn)));
        soff += 1.0f - __cosf(WQ * cmodf(dm + a * pn));
      }
      sma_l[tid]  = sa * (1.0f / 64.0f) * INV_HQ;
      sci_l[tid]  = sc2 * (1.0f / 64.0f);
      onc_l[tid]  = son * (1.0f / 64.0f);
      offc_l[tid] = soff * (1.0f / 64.0f);
    }
    if (tid < 8) {
      float s = 0;
      for (int m = 0; m < 64; ++m) s += 1.0f - __cosf(WQ * r_l[tid * 64 + m]);
      cur_l[tid] = s * (1.0f / 64.0f);
    }
    __syncthreads();

    // ---- conv embed: x[row][d] = sum_{c,m} X[c][m][row] * W[d][c*64+m] ----
    unsigned short* fbuf = scr;
    f32x8 cacc[8];
#pragma unroll
    for (int tc = 0; tc < 8; ++tc) cacc[tc] = f32x8{};
    for (int c = 0; c < 20; ++c) {
      for (int e = tid; e < 128 * 64; e += 256) {
        int row = e >> 6, m = e & 63, bi = row >> 4;
        float a  = A_l[(bi * 64 + m) * 16 + (row & 15)];
        float v = featval(c, a, b_l[bi * 64 + m], d_l[bi * 64 + m], r_l[bi * 64 + m],
                          p_l[row], sma_l[row], sci_l[row], onc_l[row], offc_l[row], cur_l[bi]);
        fbuf[row * LDF + m] = f2bf(v);
      }
      __syncthreads();
      gemm_strip(cacc, fbuf, LDF, wave * 16, 0, P.conv_wb, 1280, 0, c * 64, 2, lane);
      __syncthreads();
    }
#pragma unroll
    for (int tc = 0; tc < 8; ++tc)
#pragma unroll
      for (int v = 0; v < 8; ++v) {
        int row = wave * 16 + mbase + v, col = tc * 16 + nn;
        xf32[row * LDX + col] = cacc[tc][v] + P.conv_b[col] + P.pos[(row & 15) * 128 + col];
      }
    __syncthreads();

    // ---- encoder layers ----
    for (int l = 0; l < 2; ++l) {
      // LN1 -> abf (bf16)
      if (tid < 128) {
        float s = 0, s2 = 0;
        for (int dd = 0; dd < 128; ++dd) { float v = xf32[tid * LDX + dd]; s += v; s2 += v * v; }
        float mean = s * (1.0f / 128.0f);
        float inv = rsqrtf(s2 * (1.0f / 128.0f) - mean * mean + LN_EPS);
        for (int dd = 0; dd < 128; ++dd) {
          float v = (xf32[tid * LDX + dd] - mean) * inv * P.ln1_s[l][dd] + P.ln1_b[l][dd];
          abf[tid * LDA + dd] = f2bf(v);
        }
      }
      __syncthreads();

      // QKV GEMM -> qkvb (bf16)
      for (int blk = 0; blk < 3; ++blk) {
        f32x8 acc[8];
#pragma unroll
        for (int tc = 0; tc < 8; ++tc) acc[tc] = f32x8{};
        gemm_strip(acc, abf, LDA, wave * 16, 0, P.qkv_wb[l], 128, blk * 128, 0, 4, lane);
#pragma unroll
        for (int tc = 0; tc < 8; ++tc)
#pragma unroll
          for (int v = 0; v < 8; ++v) {
            int row = wave * 16 + mbase + v, col = blk * 128 + tc * 16 + nn;
            qkvb[row * LDQ + col] = f2bf(acc[tc][v] + P.qkv_bias[l][col]);
          }
      }
      __syncthreads();

      // attention: wave w handles batch bi = w, all 4 heads
      {
        unsigned short* sb = scr + wave * 16 * 32;  // scbuf
        int bi = wave;
        for (int h = 0; h < 4; ++h) {
          FragU fqv, fkv;
          const uint4* pq = (const uint4*)(qkvb + (bi * 16 + nn) * LDQ + h * 32 + half8);
          fqv.q[0] = pq[0]; fqv.q[1] = pq[2];
          const uint4* pk = (const uint4*)(qkvb + (bi * 16 + nn) * LDQ + 128 + h * 32 + half8);
          fkv.q[0] = pk[0]; fkv.q[1] = pk[2];
          f32x8 sc = f32x8{};
          sc = __builtin_amdgcn_wmma_f32_16x16x32_bf16(false, fqv.v, false, fkv.v,
                                                       (short)0, sc, false, false);
          // softmax across the 16 lanes of each half
#pragma unroll
          for (int v = 0; v < 8; ++v) {
            float s = sc[v] * 0.17677669529663687f;  // 1/sqrt(32)
            float mx = s;
            mx = fmaxf(mx, __shfl_xor(mx, 1, 32)); mx = fmaxf(mx, __shfl_xor(mx, 2, 32));
            mx = fmaxf(mx, __shfl_xor(mx, 4, 32)); mx = fmaxf(mx, __shfl_xor(mx, 8, 32));
            float e = __expf(s - mx);
            float sm = e;
            sm += __shfl_xor(sm, 1, 32); sm += __shfl_xor(sm, 2, 32);
            sm += __shfl_xor(sm, 4, 32); sm += __shfl_xor(sm, 8, 32);
            sc[v] = e / sm;
          }
          // write attn (zero-padded K 16..31) to per-wave LDS
#pragma unroll
          for (int v = 0; v < 8; ++v) {
            sb[(mbase + v) * 32 + nn] = f2bf(sc[v]);
            sb[(mbase + v) * 32 + 16 + nn] = 0;
          }
          FragU fat;
          const uint4* pa = (const uint4*)(sb + nn * 32 + half8);
          fat.q[0] = pa[0]; fat.q[1] = pa[2];
#pragma unroll
          for (int dh = 0; dh < 2; ++dh) {
            FragU fv;
#pragma unroll
            for (int e = 0; e < 16; ++e) {
              unsigned short val = 0;
              if (e < 8) {   // tokens half8+e (0..15) valid; padded rows are zero
                int kt = half8 + e;
                val = qkvb[(bi * 16 + kt) * LDQ + 256 + h * 32 + dh * 16 + nn];
              }
              fv.s[e] = val;
            }
            f32x8 o = f32x8{};
            o = __builtin_amdgcn_wmma_f32_16x16x32_bf16(false, fat.v, false, fv.v,
                                                        (short)0, o, false, false);
#pragma unroll
            for (int v = 0; v < 8; ++v)
              abf[(bi * 16 + mbase + v) * LDA + h * 32 + dh * 16 + nn] = f2bf(o[v]);
          }
        }
      }
      __syncthreads();

      // out-proj + residual add
      {
        f32x8 acc[8];
#pragma unroll
        for (int tc = 0; tc < 8; ++tc) acc[tc] = f32x8{};
        gemm_strip(acc, abf, LDA, wave * 16, 0, P.out_wb[l], 128, 0, 0, 4, lane);
#pragma unroll
        for (int tc = 0; tc < 8; ++tc)
#pragma unroll
          for (int v = 0; v < 8; ++v) {
            int row = wave * 16 + mbase + v, col = tc * 16 + nn;
            xf32[row * LDX + col] += acc[tc][v] + P.out_bias[l][col];
          }
      }
      __syncthreads();

      // LN2 -> abf
      if (tid < 128) {
        float s = 0, s2 = 0;
        for (int dd = 0; dd < 128; ++dd) { float v = xf32[tid * LDX + dd]; s += v; s2 += v * v; }
        float mean = s * (1.0f / 128.0f);
        float inv = rsqrtf(s2 * (1.0f / 128.0f) - mean * mean + LN_EPS);
        for (int dd = 0; dd < 128; ++dd) {
          float v = (xf32[tid * LDX + dd] - mean) * inv * P.ln2_s[l][dd] + P.ln2_b[l][dd];
          abf[tid * LDA + dd] = f2bf(v);
        }
      }
      __syncthreads();

      // fused FFN: 16 chunks of 128 hidden; per-wave rows are self-contained
      {
        unsigned short* hbf = scr;
        f32x8 facc[8];
#pragma unroll
        for (int tc = 0; tc < 8; ++tc) facc[tc] = f32x8{};
        for (int kc = 0; kc < 16; ++kc) {
          f32x8 hacc[8];
#pragma unroll
          for (int tc = 0; tc < 8; ++tc) hacc[tc] = f32x8{};
          gemm_strip(hacc, abf, LDA, wave * 16, 0, P.ff1_wb[l], 128, kc * 128, 0, 4, lane);
#pragma unroll
          for (int tc = 0; tc < 8; ++tc)
#pragma unroll
            for (int v = 0; v < 8; ++v) {
              int row = wave * 16 + mbase + v, col = tc * 16 + nn;
              float hv = hacc[tc][v] + P.ff1_bias[l][kc * 128 + col];
              hbf[row * LDH + col] = f2bf(fmaxf(hv, 0.0f));
            }
          gemm_strip(facc, hbf, LDH, wave * 16, 0, P.ff2_wb[l], 2048, 0, kc * 128, 4, lane);
        }
#pragma unroll
        for (int tc = 0; tc < 8; ++tc)
#pragma unroll
          for (int v = 0; v < 8; ++v) {
            int row = wave * 16 + mbase + v, col = tc * 16 + nn;
            xf32[row * LDX + col] += facc[tc][v] + P.ff2_bias[l][col];
          }
      }
      __syncthreads();
    }

    // ---- final LN + head + z update ----
    if (tid < 128) {
      float s = 0, s2 = 0;
      for (int dd = 0; dd < 128; ++dd) { float v = xf32[tid * LDX + dd]; s += v; s2 += v * v; }
      float mean = s * (1.0f / 128.0f);
      float inv = rsqrtf(s2 * (1.0f / 128.0f) - mean * mean + LN_EPS);
      float dz = P.head_b[0];
      for (int dd = 0; dd < 128; ++dd) {
        float v = (xf32[tid * LDX + dd] - mean) * inv * P.lnf_s[dd] + P.lnf_b[dd];
        dz += v * P.head_w[dd];
      }
      z_l[tid] += 0.5f * tanhf(dz);
    }
    __syncthreads();
  }

  if (tid < 128) P.out[(size_t)wg * 128 + tid] = z_l[tid];
}

// ---------------- host launcher ----------------
extern "C" void kernel_launch(void* const* d_in, const int* in_sizes, int n_in,
                              void* d_out, int out_size, void* d_ws, size_t ws_size,
                              hipStream_t stream) {
  (void)in_sizes; (void)n_in; (void)out_size; (void)ws_size;

  unsigned short* wsb = (unsigned short*)d_ws;
  size_t off = 0;
  unsigned short* conv_wb = wsb; off += 163840;
  unsigned short *qkv_wb[2], *out_wb[2], *ff1_wb[2], *ff2_wb[2];
  for (int l = 0; l < 2; ++l) {
    qkv_wb[l] = wsb + off; off += 49152;
    out_wb[l] = wsb + off; off += 16384;
    ff1_wb[l] = wsb + off; off += 262144;
    ff2_wb[l] = wsb + off; off += 262144;
  }

  // weight f32 -> bf16 conversions
  f2bf_kernel<<<(163840 + 255) / 256, 256, 0, stream>>>((const float*)d_in[2], conv_wb, 163840);
  for (int l = 0; l < 2; ++l) {
    int base = 5 + l * 12;
    f2bf_kernel<<<(49152 + 255) / 256, 256, 0, stream>>>((const float*)d_in[base + 2], qkv_wb[l], 49152);
    f2bf_kernel<<<(16384 + 255) / 256, 256, 0, stream>>>((const float*)d_in[base + 4], out_wb[l], 16384);
    f2bf_kernel<<<(262144 + 255) / 256, 256, 0, stream>>>((const float*)d_in[base + 8], ff1_wb[l], 262144);
    f2bf_kernel<<<(262144 + 255) / 256, 256, 0, stream>>>((const float*)d_in[base + 10], ff2_wb[l], 262144);
  }

  KParams P;
  P.A = (const float*)d_in[0];
  P.b = (const float*)d_in[1];
  P.conv_b = (const float*)d_in[3];
  P.pos = (const float*)d_in[4];
  for (int l = 0; l < 2; ++l) {
    int base = 5 + l * 12;
    P.ln1_s[l] = (const float*)d_in[base + 0];
    P.ln1_b[l] = (const float*)d_in[base + 1];
    P.qkv_bias[l] = (const float*)d_in[base + 3];
    P.out_bias[l] = (const float*)d_in[base + 5];
    P.ln2_s[l] = (const float*)d_in[base + 6];
    P.ln2_b[l] = (const float*)d_in[base + 7];
    P.ff1_bias[l] = (const float*)d_in[base + 9];
    P.ff2_bias[l] = (const float*)d_in[base + 11];
    P.qkv_wb[l] = qkv_wb[l]; P.out_wb[l] = out_wb[l];
    P.ff1_wb[l] = ff1_wb[l]; P.ff2_wb[l] = ff2_wb[l];
  }
  P.lnf_s = (const float*)d_in[29];
  P.lnf_b = (const float*)d_in[30];
  P.head_w = (const float*)d_in[31];
  P.head_b = (const float*)d_in[32];
  P.conv_wb = conv_wb;
  P.out = (float*)d_out;

  size_t floats = 8192 + 512 + 128 + 128 + 512 + 512 + 128 * 4 + 16 + 128 * LDX;
  size_t shorts = 128 * LDA + 128 * LDQ + 128 * LDH;
  size_t smemBytes = floats * 4 + shorts * 2;
  (void)hipFuncSetAttribute((const void*)vit_kernel,
                            hipFuncAttributeMaxDynamicSharedMemorySize, (int)smemBytes);

  vit_kernel<<<512, 256, smemBytes, stream>>>(P);
}
